// UNetEquiconv_55456617725963
// MI455X (gfx1250) — compile-verified, matrix operand order checked by reading, not run on hardware
//
#include <hip/hip_runtime.h>

// ---------------------------------------------------------------------------
// UNet with deformable convs on gfx1250 (MI455X).
// Every dcn = implicit GEMM (M=Co, K=Ci*kh*kw, N=Ho*Wo) using
// v_wmma_f32_16x16x32_f16. Bilinear gather is staged per 16-pixel tile into
// LDS in WMMA B-operand layout, double-buffered (1 barrier / K-step), and
// shared by 4 waves (64 output channels). All index math is 32-bit and the
// ci / k decomposition of the K index is maintained incrementally so the hot
// loop has no divisions and no 64-bit VALU address chains.
//
// Input flattening assumption (jax tree_flatten: dict keys sorted, lists in
// order) over {'x','params','offsets'}:
//   [0..29]   offsets  [30..140] params  [141] x     (see enum below)
// ---------------------------------------------------------------------------

typedef _Float16 f16;
typedef __attribute__((ext_vector_type(16))) _Float16 v16h;
typedef __attribute__((ext_vector_type(8)))  _Float16 v8h;
typedef __attribute__((ext_vector_type(8)))  float    v8f;

#define DCN_TN    16
#define DCN_KSTEP 32
#define DCN_MAXK  49   // 7x7 stem conv

// ---------------------------------------------------------------------------
// Weight pack: f32 [Co, Ci*K] -> f16 [Copad, KCpad], zero padded.
// ---------------------------------------------------------------------------
__global__ void pack_w_kernel(const float* __restrict__ w, int Co, int KC,
                              int Copad, int KCpad, f16* __restrict__ wp) {
  int i = blockIdx.x * blockDim.x + threadIdx.x;
  int total = Copad * KCpad;
  if (i >= total) return;
  int kc = i % KCpad;
  int r  = i / KCpad;
  float v = (r < Co && kc < KC) ? w[r * KC + kc] : 0.0f;
  wp[i] = (f16)v;
}

// ---------------------------------------------------------------------------
// Deformable conv as implicit GEMM with WMMA.
//   grid.x : pixel tiles of 16 (flattened Ho*Wo)
//   grid.y : blocks of 64 output channels (4 waves x 16)
//   block  : 128 threads = 4 wave32
// ---------------------------------------------------------------------------
__global__ __launch_bounds__(128) void dcn_wmma_kernel(
    const float* __restrict__ src, int Ci, int H, int W,
    const f16* __restrict__ wpack, int KCpad, int Copad,
    const float* __restrict__ offs,
    const float* __restrict__ bias,
    const float* __restrict__ bng, const float* __restrict__ bnb,
    const float* __restrict__ bnm, const float* __restrict__ bnv,
    int relu,
    int Co, int Ho, int Wo, int kh, int kw, int stride, int pad,
    float* __restrict__ dst) {
  // sampling table: one 32B row per (k, n): {o00,o01,o10,o11, w00,w01,w10,w11}
  __shared__ __align__(16) int sTab[DCN_MAXK * DCN_TN][8];
  // double-buffered B tile in WMMA B-operand register layout
  __shared__ __align__(32) f16 bbuf[2][DCN_KSTEP * DCN_TN];

  const int tid  = threadIdx.x;
  const int lane = tid & 31;
  const int wave = tid >> 5;
  const int half = lane >> 4;
  const int K    = kh * kw;
  const int KC   = Ci * K;
  const int HW   = H * W;
  const int HoWo = Ho * Wo;
  const int pixBase = blockIdx.x * DCN_TN;

  // ---- phase 0: per-(k, pixel) bilinear sampling params (independent of ci)
  for (int idx = tid; idx < K * DCN_TN; idx += 128) {
    int k = idx >> 4;
    int n = idx & 15;
    int pixel = pixBase + n;
    float w00 = 0.f, w01 = 0.f, w10 = 0.f, w11 = 0.f;
    int o00 = 0, o01 = 0, o10 = 0, o11 = 0;
    if (pixel < HoWo) {
      int ho = pixel / Wo, wo = pixel - ho * Wo;
      int ky = k / kw, kx = k - ky * kw;
      float dy = offs[(2 * k) * HoWo + pixel];
      float dx = offs[(2 * k + 1) * HoWo + pixel];
      float py = (float)(ho * stride - pad + ky) + dy;
      float px = (float)(wo * stride - pad + kx) + dx;
      float y0f = floorf(py), x0f = floorf(px);
      float wy = py - y0f, wx = px - x0f;
      int y0 = (int)y0f, x0 = (int)x0f;
      int y1 = y0 + 1, x1 = x0 + 1;
      bool my0 = (y0 >= 0) && (y0 < H), my1 = (y1 >= 0) && (y1 < H);
      bool mx0 = (x0 >= 0) && (x0 < W), mx1 = (x1 >= 0) && (x1 < W);
      int y0c = min(max(y0, 0), H - 1), y1c = min(max(y1, 0), H - 1);
      int x0c = min(max(x0, 0), W - 1), x1c = min(max(x1, 0), W - 1);
      o00 = y0c * W + x0c; o01 = y0c * W + x1c;
      o10 = y1c * W + x0c; o11 = y1c * W + x1c;
      w00 = (my0 && mx0) ? (1.f - wy) * (1.f - wx) : 0.f;
      w01 = (my0 && mx1) ? (1.f - wy) * wx : 0.f;
      w10 = (my1 && mx0) ? wy * (1.f - wx) : 0.f;
      w11 = (my1 && mx1) ? wy * wx : 0.f;
    }
    int* t = sTab[idx];
    t[0] = o00; t[1] = o01; t[2] = o10; t[3] = o11;
    t[4] = __float_as_int(w00); t[5] = __float_as_int(w01);
    t[6] = __float_as_int(w10); t[7] = __float_as_int(w11);
  }

  // ---- per-thread gather state: fixed (kcl, n0..n0+3); ci/k kept incrementally
  const int kcl = (tid * 4) >> 4;   // same for all 4 elements of this thread
  const int n0  = (tid * 4) & 15;
  int gk, gciHW;
  {
    int ci = kcl / K;               // one division, outside the loop
    gk = kcl - ci * K;
    gciHW = ci * HW;
  }
  const int dCi   = DCN_KSTEP / K;
  const int dCiHW = dCi * HW;
  const int dK    = DCN_KSTEP - dCi * K;   // == KSTEP % K
  // fixed LDS store slot (B-operand layout): lane = n + 16*((kcl>>3)&1),
  // elem = (kcl&16 ? 8:0) + (kcl&7); slot_e = slot0 + 16*e
  const int hb16  = ((kcl >> 3) & 1) << 4;
  const int be    = ((kcl & 16) ? 8 : 0) + (kcl & 7);
  const int slot0 = (n0 + hb16) * 16 + be;

  auto gather = [&](int kc0, f16* bb) {
    bool valid = (kc0 + kcl) < KC;
    const int* t = &sTab[gk * DCN_TN + n0][0];
#pragma unroll
    for (int e = 0; e < 4; ++e) {
      float v = 0.f;
      if (valid) {
        const int* te = t + e * 8;           // two ds_load_b128 per element
        int   o0 = te[0], o1 = te[1], o2 = te[2], o3 = te[3];
        float w0 = __int_as_float(te[4]), w1 = __int_as_float(te[5]);
        float w2 = __int_as_float(te[6]), w3 = __int_as_float(te[7]);
        v = w0 * src[gciHW + o0] + w1 * src[gciHW + o1] +
            w2 * src[gciHW + o2] + w3 * src[gciHW + o3];
      }
      bb[slot0 + e * 16] = (f16)v;
    }
    gciHW += dCiHW;                          // advance kc by KSTEP
    gk    += dK;
    if (gk >= K) { gk -= K; gciHW += HW; }
  };

  __syncthreads();            // sTab ready
  gather(0, bbuf[0]);         // prologue tile
  __syncthreads();

  // A fragment row for this wave's 16-channel tile (clamped: pad rows are 0)
  int rowA = min(blockIdx.y * 64 + wave * 16 + (lane & 15), Copad - 1);
  const f16* wrow = wpack + rowA * KCpad;

  v8f acc = {0.f, 0.f, 0.f, 0.f, 0.f, 0.f, 0.f, 0.f};
  int cur = 0;
  for (int kc0 = 0; kc0 < KCpad; kc0 += DCN_KSTEP) {
    // A fragment: two contiguous 16B loads from packed weights
    v8h a0 = *(const v8h*)(wrow + kc0 + half * 8);
    v8h a1 = *(const v8h*)(wrow + kc0 + 16 + half * 8);
    // B fragment from current LDS buffer
    const f16* bp = &bbuf[cur][lane * 16];
    v8h b0 = *(const v8h*)(bp);
    v8h b1 = *(const v8h*)(bp + 8);
    // overlap: gather next tile into the other buffer (uniform condition)
    if (kc0 + DCN_KSTEP < KCpad) gather(kc0 + DCN_KSTEP, bbuf[cur ^ 1]);
    v16h a, b;
#pragma unroll
    for (int i = 0; i < 8; ++i) {
      a[i] = a0[i]; a[i + 8] = a1[i];
      b[i] = b0[i]; b[i + 8] = b1[i];
    }
    acc = __builtin_amdgcn_wmma_f32_16x16x32_f16(false, a, false, b,
                                                 (short)0, acc, false, false);
    __syncthreads();          // one barrier per K-step
    cur ^= 1;
  }

  // ---- epilogue: C layout -> (co = r + 8*half, n = lane&15); bias/BN/ReLU
  int n = lane & 15;
  int pixel = pixBase + n;
  if (pixel < HoWo) {
#pragma unroll
    for (int r = 0; r < 8; ++r) {
      int co = blockIdx.y * 64 + wave * 16 + half * 8 + r;
      if (co < Co) {
        float v = acc[r];
        if (bias) v += bias[co];
        if (bng) {
          float inv = rsqrtf(bnv[co] + 1e-5f);
          v = (v - bnm[co]) * inv * bng[co] + bnb[co];
        }
        if (relu) v = fmaxf(v, 0.f);
        dst[co * HoWo + pixel] = v;
      }
    }
  }
}

// ---------------------------------------------------------------------------
// 3x3 / stride-2 / pad-1 max pool
// ---------------------------------------------------------------------------
__global__ void maxpool3s2_kernel(const float* __restrict__ src, int C, int H,
                                  int W, int Ho, int Wo, float* __restrict__ dst) {
  int i = blockIdx.x * blockDim.x + threadIdx.x;
  int total = C * Ho * Wo;
  if (i >= total) return;
  int wo = i % Wo;
  int t = i / Wo;
  int ho = t % Ho;
  int c = t / Ho;
  float m = -INFINITY;
  int y0 = ho * 2 - 1, x0 = wo * 2 - 1;
  for (int dy = 0; dy < 3; ++dy)
    for (int dx = 0; dx < 3; ++dx) {
      int y = y0 + dy, x = x0 + dx;
      if (y >= 0 && y < H && x >= 0 && x < W)
        m = fmaxf(m, src[c * H * W + y * W + x]);
    }
  dst[i] = m;
}

// ---------------------------------------------------------------------------
// Bilinear x2 upsample, align_corners=True. dst is the first C channels of a
// (possibly larger) concat buffer -> contiguous [C, 2H, 2W] region.
// ---------------------------------------------------------------------------
__global__ void up2_kernel(const float* __restrict__ src, int C, int H, int W,
                           float* __restrict__ dst) {
  int Ho = 2 * H, Wo = 2 * W;
  int i = blockIdx.x * blockDim.x + threadIdx.x;
  int total = C * Ho * Wo;
  if (i >= total) return;
  int wo = i % Wo;
  int t = i / Wo;
  int ho = t % Ho;
  int c = t / Ho;
  float sy = (H > 1) ? (float)(H - 1) / (float)(Ho - 1) : 0.f;
  float sx = (W > 1) ? (float)(W - 1) / (float)(Wo - 1) : 0.f;
  float cy = ho * sy, cx = wo * sx;
  int y0 = (int)floorf(cy), x0 = (int)floorf(cx);
  int y1 = min(y0 + 1, H - 1), x1 = min(x0 + 1, W - 1);
  float ty = cy - (float)y0, tx = cx - (float)x0;
  const float* b = src + c * H * W;
  float v = (1.f - ty) * ((1.f - tx) * b[y0 * W + x0] + tx * b[y0 * W + x1]) +
            ty * ((1.f - tx) * b[y1 * W + x0] + tx * b[y1 * W + x1]);
  dst[i] = v;
}

// ---------------------------------------------------------------------------
// Input index table (see flattening note at top)
// ---------------------------------------------------------------------------
enum {
  O_L0 = 0, O_L0_1X1 = 1, O_L1_0 = 2, O_L1_1X1 = 6,
  O_L2_0 = 7, O_L2_1X1 = 11, O_L3_0 = 12, O_L3_1X1 = 16,
  O_L4_0 = 17, O_L4_1X1 = 21, O_LAST = 22,
  O_ORIG0 = 23, O_ORIG1 = 24, O_ORIG2 = 25,
  O_UP0 = 26, O_UP1 = 27, O_UP2 = 28, O_UP3 = 29,
  P_L0_BN_B = 30, P_L0_BN_G = 31, P_L0_BN_M = 32, P_L0_BN_V = 33, P_L0_W = 34,
  P_L0_1X1_B = 35, P_L0_1X1_W = 36,
  P_L1_BASE = 37,                       // 4 x {bn.b,bn.g,bn.m,bn.v,w}
  P_L1_1X1_B = 57, P_L1_1X1_W = 58,
  P_L2_BASE = 59,
  P_L2_1X1_B = 79, P_L2_1X1_W = 80,
  P_L3_BASE = 81,
  P_L3_1X1_B = 101, P_L3_1X1_W = 102,
  P_L4_BASE = 103,
  P_L4_1X1_B = 123, P_L4_1X1_W = 124,
  P_LAST_B = 125, P_LAST_W = 126,
  P_ORIG0_B = 127, P_ORIG0_W = 128,
  P_ORIG1_B = 129, P_ORIG1_W = 130,
  P_ORIG2_B = 131, P_ORIG2_W = 132,
  P_UP0_B = 133, P_UP0_W = 134,
  P_UP1_B = 135, P_UP1_W = 136,
  P_UP2_B = 137, P_UP2_W = 138,
  P_UP3_B = 139, P_UP3_W = 140,
  IN_X = 141,
  N_EXPECTED = 142
};

extern "C" void kernel_launch(void* const* d_in, const int* in_sizes, int n_in,
                              void* d_out, int out_size, void* d_ws,
                              size_t ws_size, hipStream_t stream) {
  (void)in_sizes; (void)out_size; (void)ws_size;
  if (n_in < N_EXPECTED) return;

  auto GIN = [&](int i) -> const float* { return (const float*)d_in[i]; };

  // --- deterministic bump allocator over workspace ---
  char* wsbase = (char*)d_ws;
  size_t wsoff = 0;
  auto allocF = [&](size_t n) -> float* {
    float* p = (float*)(wsbase + wsoff);
    wsoff += ((n * sizeof(float)) + 255) & ~(size_t)255;
    return p;
  };
  auto allocH = [&](size_t n) -> f16* {
    f16* p = (f16*)(wsbase + wsoff);
    wsoff += ((n * sizeof(f16)) + 255) & ~(size_t)255;
    return p;
  };

  auto pack = [&](int widx, int Co, int Ci, int K) -> f16* {
    int KC = Ci * K, KCpad = (KC + 31) & ~31, Copad = (Co + 15) & ~15;
    f16* wp = allocH((size_t)Copad * KCpad);
    int total = Copad * KCpad;
    pack_w_kernel<<<dim3((total + 255) / 256), dim3(256), 0, stream>>>(
        GIN(widx), Co, KC, Copad, KCpad, wp);
    return wp;
  };

  auto dcn = [&](const float* src, int Ci, int H, int W, const f16* wp, int Co,
                 int kh, int kw, int stride, int pad, int offIdx,
                 const float* bias, const float* g, const float* bb,
                 const float* m, const float* vv, int relu, float* dst) {
    int Ho = (H + 2 * pad - kh) / stride + 1;
    int Wo = (W + 2 * pad - kw) / stride + 1;
    int KC = Ci * kh * kw, KCpad = (KC + 31) & ~31, Copad = (Co + 15) & ~15;
    dim3 grid((Ho * Wo + DCN_TN - 1) / DCN_TN, (Copad + 63) / 64);
    dcn_wmma_kernel<<<grid, dim3(128), 0, stream>>>(
        src, Ci, H, W, wp, KCpad, Copad, GIN(offIdx), bias, g, bb, m, vv, relu,
        Co, Ho, Wo, kh, kw, stride, pad, dst);
  };

  auto up2go = [&](const float* src, int C, int H, int W, float* dst) {
    int total = C * 4 * H * W;
    up2_kernel<<<dim3((total + 255) / 256), dim3(256), 0, stream>>>(src, C, H, W, dst);
  };

  // --- pack all weights into f16 GEMM layout ---
  f16* W_orig0 = pack(P_ORIG0_W, 64, 3, 9);
  f16* W_orig1 = pack(P_ORIG1_W, 64, 64, 9);
  f16* W_l0    = pack(P_L0_W, 64, 3, 49);
  f16 *W_l1[4], *W_l2[4], *W_l3[4], *W_l4[4];
  for (int i = 0; i < 4; ++i) W_l1[i] = pack(P_L1_BASE + 5 * i + 4, 64, 64, 9);
  for (int i = 0; i < 4; ++i) W_l2[i] = pack(P_L2_BASE + 5 * i + 4, 128, i == 0 ? 64 : 128, 9);
  for (int i = 0; i < 4; ++i) W_l3[i] = pack(P_L3_BASE + 5 * i + 4, 256, i == 0 ? 128 : 256, 9);
  for (int i = 0; i < 4; ++i) W_l4[i] = pack(P_L4_BASE + 5 * i + 4, 512, i == 0 ? 256 : 512, 9);
  f16* W_l41 = pack(P_L4_1X1_W, 512, 512, 1);
  f16* W_l31 = pack(P_L3_1X1_W, 256, 256, 1);
  f16* W_l21 = pack(P_L2_1X1_W, 128, 128, 1);
  f16* W_l11 = pack(P_L1_1X1_W, 64, 64, 1);
  f16* W_l01 = pack(P_L0_1X1_W, 64, 64, 1);
  f16* W_up3 = pack(P_UP3_W, 512, 768, 9);
  f16* W_up2w = pack(P_UP2_W, 256, 640, 9);
  f16* W_up1w = pack(P_UP1_W, 256, 320, 9);
  f16* W_up0w = pack(P_UP0_W, 128, 320, 9);
  f16* W_or2 = pack(P_ORIG2_W, 64, 192, 9);
  f16* W_last = pack(P_LAST_W, 13, 64, 1);

  // --- activation / concat buffers (channels contiguous => free concat) ---
  float* xo0 = allocF((size_t)64 * 32768);
  float* l0b = allocF((size_t)64 * 8192);
  float* mpb = allocF((size_t)64 * 2048);
  float* s1a = allocF((size_t)64 * 2048);  float* s1b = allocF((size_t)64 * 2048);
  float* s2a = allocF((size_t)128 * 512);  float* s2b = allocF((size_t)128 * 512);
  float* s3a = allocF((size_t)256 * 128);  float* s3b = allocF((size_t)256 * 128);
  float* s4a = allocF((size_t)512 * 32);   float* s4b = allocF((size_t)512 * 32);
  float* u4 = allocF((size_t)512 * 32);
  float* C3 = allocF((size_t)768 * 128);
  float* h3 = allocF((size_t)512 * 128);
  float* C2 = allocF((size_t)640 * 512);
  float* h2 = allocF((size_t)256 * 512);
  float* C1 = allocF((size_t)320 * 2048);
  float* h1 = allocF((size_t)256 * 2048);
  float* C0 = allocF((size_t)320 * 8192);
  float* h0 = allocF((size_t)128 * 8192);
  float* CX = allocF((size_t)192 * 32768);
  float* hF = allocF((size_t)64 * 32768);

  const float* x = GIN(IN_X);

  // --- stem / orig branch ---
  dcn(x, 3, 128, 256, W_orig0, 64, 3, 3, 1, 1, O_ORIG0, GIN(P_ORIG0_B),
      nullptr, nullptr, nullptr, nullptr, 1, xo0);
  // orig1 output goes straight into final concat buffer (channels 128..191)
  dcn(xo0, 64, 128, 256, W_orig1, 64, 3, 3, 1, 1, O_ORIG1, GIN(P_ORIG1_B),
      nullptr, nullptr, nullptr, nullptr, 1, CX + (size_t)128 * 32768);
  dcn(x, 3, 128, 256, W_l0, 64, 7, 7, 2, 3, O_L0, nullptr, GIN(P_L0_BN_G),
      GIN(P_L0_BN_B), GIN(P_L0_BN_M), GIN(P_L0_BN_V), 1, l0b);
  {
    int total = 64 * 32 * 64;
    maxpool3s2_kernel<<<dim3((total + 255) / 256), dim3(256), 0, stream>>>(
        l0b, 64, 64, 128, 32, 64, mpb);
  }

  // --- stage l1 (stride 1) ---
  {
    const float* in = mpb;
    float* bufs[2] = {s1a, s1b};
    for (int i = 0; i < 4; ++i) {
      int base = P_L1_BASE + 5 * i;
      dcn(in, 64, 32, 64, W_l1[i], 64, 3, 3, 1, 1, O_L1_0 + i, nullptr,
          GIN(base + 1), GIN(base + 0), GIN(base + 2), GIN(base + 3),
          (i % 2 == 0) ? 1 : 0, bufs[i & 1]);
      in = bufs[i & 1];
    }
  }
  const float* l1 = s1b;

  // --- stages l2..l4 (first block stride 2) ---
  auto stage2 = [&](const float* in0, int Ci0, int C, int Hin, int Win,
                    int pBase, int oBase, f16* const* Wb, float* bufA,
                    float* bufB) -> const float* {
    int Ho = (Hin - 1) / 2 + 1, Wo = (Win - 1) / 2 + 1;
    float* bufs[2] = {bufA, bufB};
    const float* in = in0;
    for (int i = 0; i < 4; ++i) {
      int base = pBase + 5 * i;
      dcn(in, i == 0 ? Ci0 : C, i == 0 ? Hin : Ho, i == 0 ? Win : Wo, Wb[i], C,
          3, 3, i == 0 ? 2 : 1, 1, oBase + i, nullptr, GIN(base + 1),
          GIN(base + 0), GIN(base + 2), GIN(base + 3), (i % 2 == 0) ? 1 : 0,
          bufs[i & 1]);
      in = bufs[i & 1];
    }
    return in;
  };
  const float* l2 = stage2(l1, 64, 128, 32, 64, P_L2_BASE, O_L2_0, W_l2, s2a, s2b);
  const float* l3 = stage2(l2, 128, 256, 16, 32, P_L3_BASE, O_L3_0, W_l3, s3a, s3b);
  const float* l4 = stage2(l3, 256, 512, 8, 16, P_L4_BASE, O_L4_0, W_l4, s4a, s4b);

  // --- decoder: upsample + lateral 1x1 into concat buffers, then 3x3 dcn ---
  dcn(l4, 512, 4, 8, W_l41, 512, 1, 1, 1, 0, O_L4_1X1, GIN(P_L4_1X1_B),
      nullptr, nullptr, nullptr, nullptr, 1, u4);
  up2go(u4, 512, 4, 8, C3);
  dcn(l3, 256, 8, 16, W_l31, 256, 1, 1, 1, 0, O_L3_1X1, GIN(P_L3_1X1_B),
      nullptr, nullptr, nullptr, nullptr, 1, C3 + (size_t)512 * 128);
  dcn(C3, 768, 8, 16, W_up3, 512, 3, 3, 1, 1, O_UP3, GIN(P_UP3_B),
      nullptr, nullptr, nullptr, nullptr, 1, h3);

  up2go(h3, 512, 8, 16, C2);
  dcn(l2, 128, 16, 32, W_l21, 128, 1, 1, 1, 0, O_L2_1X1, GIN(P_L2_1X1_B),
      nullptr, nullptr, nullptr, nullptr, 1, C2 + (size_t)512 * 512);
  dcn(C2, 640, 16, 32, W_up2w, 256, 3, 3, 1, 1, O_UP2, GIN(P_UP2_B),
      nullptr, nullptr, nullptr, nullptr, 1, h2);

  up2go(h2, 256, 16, 32, C1);
  dcn(l1, 64, 32, 64, W_l11, 64, 1, 1, 1, 0, O_L1_1X1, GIN(P_L1_1X1_B),
      nullptr, nullptr, nullptr, nullptr, 1, C1 + (size_t)256 * 2048);
  dcn(C1, 320, 32, 64, W_up1w, 256, 3, 3, 1, 1, O_UP1, GIN(P_UP1_B),
      nullptr, nullptr, nullptr, nullptr, 1, h1);

  up2go(h1, 256, 32, 64, C0);
  dcn(l0b, 64, 64, 128, W_l01, 64, 1, 1, 1, 0, O_L0_1X1, GIN(P_L0_1X1_B),
      nullptr, nullptr, nullptr, nullptr, 1, C0 + (size_t)256 * 8192);
  dcn(C0, 320, 64, 128, W_up0w, 128, 3, 3, 1, 1, O_UP0, GIN(P_UP0_B),
      nullptr, nullptr, nullptr, nullptr, 1, h0);

  up2go(h0, 128, 64, 128, CX);  // channels 0..127; 128..191 = orig branch
  dcn(CX, 192, 128, 256, W_or2, 64, 3, 3, 1, 1, O_ORIG2, GIN(P_ORIG2_B),
      nullptr, nullptr, nullptr, nullptr, 1, hF);

  // --- head: 1x1 dcn to 13 classes, no relu, straight into d_out ---
  dcn(hF, 64, 128, 256, W_last, 13, 1, 1, 1, 0, O_LAST, GIN(P_LAST_B),
      nullptr, nullptr, nullptr, nullptr, 0, (float*)d_out);
}
